// SE3Transformer_7387343749390
// MI455X (gfx1250) — compile-verified, matrix operand order checked by reading, not run on hardware
//
#include <hip/hip_runtime.h>
#include <math.h>

// ---------------------------------------------------------------------------
// SE(3)-Transformer forward for MI455X (gfx1250, wave32, WMMA).
// Heavy GEMM (radial W3 @ hidden, per-edge) runs on v_wmma_f32_16x16x32_bf16,
// fused with the basis/t contraction so R (up to 235MB/pair) never touches HBM.
// ---------------------------------------------------------------------------

typedef __attribute__((ext_vector_type(16))) __bf16 v16bf;
typedef __attribute__((ext_vector_type(8)))  float  v8f;

constexpr int Nn = 4096;
constexpr int Ee = 32768;
constexpr int Gg = 64;

static __device__ __forceinline__ void atomicMaxF(float* a, float v) {
  if (v >= 0.f) atomicMax((int*)a, __float_as_int(v));
  else          atomicMin((unsigned int*)a, __float_as_uint(v));
}

// ------------------------------ small utility kernels ----------------------

__global__ void k_fill(float* __restrict__ p, float v, int n) {
  int i = blockIdx.x * blockDim.x + threadIdx.x;
  if (i < n) p[i] = v;
}

__global__ void k_copy(const float* __restrict__ s, float* __restrict__ d, int n) {
  int i = blockIdx.x * blockDim.x + threadIdx.x;
  if (i < n) d[i] = s[i];
}

__global__ void k_scale(float* __restrict__ p, float s, int n) {
  int i = blockIdx.x * blockDim.x + threadIdx.x;
  if (i < n) p[i] *= s;
}

// xe = concat(r, edge_feat)  ->  (E,5)
__global__ void k_build_xe(const float* __restrict__ r, const float* __restrict__ ef,
                           float* __restrict__ xe) {
  int e = blockIdx.x * blockDim.x + threadIdx.x;
  if (e >= Ee) return;
  xe[e * 5 + 0] = r[e];
#pragma unroll
  for (int j = 0; j < 4; ++j) xe[e * 5 + 1 + j] = ef[e * 4 + j];
}

// ------------------------------ radial MLP hidden --------------------------
// h2 = relu(relu(xe@W1+b1)@W2+b2), per edge. (tiny: 1.2K MAC/edge)
__global__ __launch_bounds__(128)
void k_radial_hidden(const float* __restrict__ xe,
                     const float* __restrict__ W1, const float* __restrict__ b1,
                     const float* __restrict__ W2, const float* __restrict__ b2,
                     float* __restrict__ h2) {
  __shared__ float W1s[5 * 32], b1s[32], W2s[32 * 32], b2s[32];
  for (int i = threadIdx.x; i < 5 * 32; i += 128) W1s[i] = W1[i];
  for (int i = threadIdx.x; i < 32 * 32; i += 128) W2s[i] = W2[i];
  if (threadIdx.x < 32) { b1s[threadIdx.x] = b1[threadIdx.x]; b2s[threadIdx.x] = b2[threadIdx.x]; }
  __syncthreads();
  int e = blockIdx.x * 128 + threadIdx.x;
  if (e >= Ee) return;
  float x[5];
#pragma unroll
  for (int i = 0; i < 5; ++i) x[i] = xe[e * 5 + i];
  float h1[32];
#pragma unroll
  for (int k = 0; k < 32; ++k) {
    float a = b1s[k];
#pragma unroll
    for (int i = 0; i < 5; ++i) a += x[i] * W1s[i * 32 + k];
    h1[k] = fmaxf(a, 0.f);
  }
#pragma unroll
  for (int k = 0; k < 32; ++k) {
    float a = b2s[k];
#pragma unroll
    for (int j = 0; j < 32; ++j) a += h1[j] * W2s[j * 32 + k];
    h2[(size_t)e * 32 + k] = fmaxf(a, 0.f);
  }
}

// ------------------------------ fused pair-conv (WMMA) ---------------------
// One wave per 16-edge tile:
//   t[e,d,m,f]  = sum_n basis[e,m,n,f] * hsrc[src(e),d,n]           (LDS, bf16)
//   R tile      = h2tile(16x32) @ W3(32xK)   via v_wmma_f32_16x16x32_bf16
//   msg[e,c,m] += sum_{d,f} R[e,(c,d,f)] * t[e,d,m,f]               (ds_add_f32)
template <int DI, int DO, int NF, int MO>
__global__ __launch_bounds__(32)
void k_conv_pair(const float* __restrict__ h2,     // (E,32)
                 const float* __restrict__ W3,     // (32, K)
                 const float* __restrict__ basis,  // (E, 2DO+1, 2DI+1, NF)
                 const float* __restrict__ hsrc,   // (N, 32, 2DI+1)
                 const int*   __restrict__ esrc,
                 float*       __restrict__ msg) {  // (E, MO, 2DO+1) accumulated
  constexpr int MI  = 32;
  constexpr int M_O = 2 * DO + 1;
  constexpr int M_I = 2 * DI + 1;
  constexpr int K   = MO * MI * NF;
  constexpr int TN  = 16;
  static_assert(K % 16 == 0, "K must be multiple of 16");

  __shared__ __bf16 srcS[TN * MI * M_I];
  __shared__ __bf16 tS[TN * MI * M_O * NF];
  __shared__ float  msgS[TN * MO * M_O];

  const int e0    = blockIdx.x * TN;
  const int lane  = threadIdx.x;       // 0..31 (wave32)
  const int l16   = lane & 15;
  const int khalf = lane >> 4;         // 0/1 : K half per ISA 16-bit A/B layout

  // --- A fragment: 16 edges x 32 hidden (bf16), ISA 16x32 16-bit A packing
  v16bf a;
  {
    const float* hrow = h2 + (size_t)(e0 + l16) * 32;
#pragma unroll
    for (int i = 0; i < 16; ++i) {
      int k = (i < 8) ? (khalf * 8 + i) : (16 + khalf * 8 + (i - 8));
      a[i] = (__bf16)hrow[k];
    }
  }

  // --- stage gathered source features
  for (int idx = lane; idx < TN * MI * M_I; idx += 32) {
    int e = idx / (MI * M_I);
    int r = idx % (MI * M_I);
    int s = esrc[e0 + e];
    srcS[idx] = (__bf16)hsrc[(size_t)s * MI * M_I + r];
  }
  __syncthreads();

  // --- build t in LDS
  for (int idx = lane; idx < TN * MI * M_O * NF; idx += 32) {
    int e   = idx / (MI * M_O * NF);
    int rem = idx % (MI * M_O * NF);
    int d   = rem / (M_O * NF);
    int mf  = rem % (M_O * NF);
    int m   = mf / NF;
    int f   = mf % NF;
    const float*  bp = basis + ((size_t)(e0 + e) * M_O + m) * (M_I * NF) + f;
    const __bf16* sp = srcS + (e * MI + d) * M_I;
    float acc = 0.f;
#pragma unroll
    for (int n = 0; n < M_I; ++n) acc += bp[n * NF] * (float)sp[n];
    tS[idx] = (__bf16)acc;
  }
  for (int idx = lane; idx < TN * MO * M_O; idx += 32) msgS[idx] = 0.f;
  __syncthreads();

  // --- sweep B tiles of W3, WMMA, immediately contract with t
  for (int j0 = 0; j0 < K; j0 += 16) {
    const int j = j0 + l16;            // this lane's column (N of B / C)
    v16bf b;
#pragma unroll
    for (int i = 0; i < 16; ++i) {
      int k = (i < 8) ? (khalf * 8 + i) : (16 + khalf * 8 + (i - 8));
      b[i] = (__bf16)W3[(size_t)k * K + j];
    }
    v8f c = {};
    c = __builtin_amdgcn_wmma_f32_16x16x32_bf16(false, a, false, b,
                                                (short)0, c, false, false);
    // column j -> (c_idx, d_idx, f_idx); rows (VGPR g) -> edges
    const int ci  = j / (MI * NF);
    const int rem = j % (MI * NF);
    const int dd  = rem / NF;
    const int ff  = rem % NF;
#pragma unroll
    for (int g = 0; g < 8; ++g) {
      const int e   = g + 8 * khalf;   // C/D layout: M = g + 8*(lane/16)
      const float v = c[g];
#pragma unroll
      for (int m = 0; m < M_O; ++m) {
        float tv = (float)tS[((e * MI + dd) * M_O + m) * NF + ff];
        atomicAdd(&msgS[(e * MO + ci) * M_O + m], v * tv);
      }
    }
  }
  __syncthreads();
  for (int idx = lane; idx < TN * MO * M_O; idx += 32) {
    int e = idx / (MO * M_O);
    int r = idx % (MO * M_O);
    msg[(size_t)(e0 + e) * MO * M_O + r] += msgS[idx];
  }
}

// ------------------------------ attention pieces ---------------------------

// q[n,o,m] = sum_k Wq[o,k] * h[n,k,m]
__global__ void k_q(const float* __restrict__ Wq, const float* __restrict__ h,
                    float* __restrict__ q, int M, int total) {
  int i = blockIdx.x * blockDim.x + threadIdx.x;
  if (i >= total) return;
  int m = i % M, t = i / M, o = t % 8, n = t / 8;
  float a = 0.f;
  for (int k = 0; k < 32; ++k) a += Wq[o * 32 + k] * h[((size_t)n * 32 + k) * M + m];
  q[i] = a;
}

// logits[e,h] += sum_{c in head, m} k[e,c,m] * q[dst(e),c,m]
__global__ void k_logits(const float* __restrict__ kb, const float* __restrict__ qb,
                         const int* __restrict__ dst, float* __restrict__ logits, int M) {
  int i = blockIdx.x * blockDim.x + threadIdx.x;
  if (i >= Ee * 2) return;
  int h = i & 1, e = i >> 1;
  int d = dst[e];
  float a = 0.f;
  for (int cc = 0; cc < 4; ++cc) {
    int c = h * 4 + cc;
    for (int m = 0; m < M; ++m)
      a += kb[((size_t)e * 8 + c) * M + m] * qb[((size_t)d * 8 + c) * M + m];
  }
  logits[i] += a;
}

__global__ void k_segmax(const float* __restrict__ l, const int* __restrict__ dst,
                         float* __restrict__ mx) {
  int i = blockIdx.x * blockDim.x + threadIdx.x;
  if (i >= Ee * 2) return;
  int h = i & 1, e = i >> 1;
  atomicMaxF(&mx[dst[e] * 2 + h], l[i]);
}

__global__ void k_expsum(float* __restrict__ l, const int* __restrict__ dst,
                         const float* __restrict__ mx, float* __restrict__ den) {
  int i = blockIdx.x * blockDim.x + threadIdx.x;
  if (i >= Ee * 2) return;
  int h = i & 1, e = i >> 1;
  float v = __expf(l[i] - mx[dst[e] * 2 + h]);
  l[i] = v;                               // logits buffer becomes ex/alpha
  atomicAdd(&den[dst[e] * 2 + h], v);
}

__global__ void k_alpha(float* __restrict__ l, const int* __restrict__ dst,
                        const float* __restrict__ den) {
  int i = blockIdx.x * blockDim.x + threadIdx.x;
  if (i >= Ee * 2) return;
  int h = i & 1, e = i >> 1;
  l[i] /= (den[dst[e] * 2 + h] + 1e-9f);
}

// agg[dst,c,m] += alpha[e,head(c)] * v[e,c,m]
__global__ void k_aggv(const float* __restrict__ v, const float* __restrict__ alpha,
                       const int* __restrict__ dst, float* __restrict__ agg,
                       int M, int total) {
  int i = blockIdx.x * blockDim.x + threadIdx.x;
  if (i >= total) return;
  int m = i % M, t = i / M, c = t % 8, e = t / 8;
  atomicAdd(&agg[((size_t)dst[e] * 8 + c) * M + m], alpha[e * 2 + (c >> 2)] * v[i]);
}

// out[n,o,m] = sum_c P[o,c]*agg[n,c,m] (+ residual)
__global__ void k_proj(const float* __restrict__ P, const float* __restrict__ agg,
                       const float* __restrict__ hres, float* __restrict__ out,
                       int M, int total) {
  int i = blockIdx.x * blockDim.x + threadIdx.x;
  if (i >= total) return;
  int m = i % M, t = i / M, o = t % 32, n = t / 32;
  float a = hres ? hres[i] : 0.f;
#pragma unroll
  for (int c = 0; c < 8; ++c) a += P[o * 8 + c] * agg[((size_t)n * 8 + c) * M + m];
  out[i] = a;
}

// gated norm per node/degree: 1 block per node, lane = channel
__global__ __launch_bounds__(32)
void k_gnorm(const float* __restrict__ g, const float* __restrict__ beta,
             const float* __restrict__ W, const float* __restrict__ b,
             const float* __restrict__ x, float* __restrict__ out, int M) {
  int n = blockIdx.x, c = threadIdx.x;
  __shared__ float aS[32], nS[32];
  const float* xr = x + ((size_t)n * 32 + c) * M;
  float ss = 0.f;
  for (int m = 0; m < M; ++m) ss += xr[m] * xr[m];
  float norm = sqrtf(fmaxf(ss, 1e-12f));
  nS[c] = norm;
  aS[c] = fmaxf(norm * g[c] + beta[c], 0.f);
  __syncthreads();
  float s = b[c];
#pragma unroll
  for (int j = 0; j < 32; ++j) s += aS[j] * W[j * 32 + c];
  float inv = s / nS[c];
  float* orow = out + ((size_t)n * 32 + c) * M;
  for (int m = 0; m < M; ++m) orow[m] = xr[m] * inv;
}

// ------------------------------ final head ---------------------------------

__global__ void k_faggr(const float* __restrict__ msg0, const int* __restrict__ dst,
                        float* __restrict__ aggF) {
  int i = blockIdx.x * blockDim.x + threadIdx.x;
  if (i >= Ee * 128) return;
  int c = i % 128, e = i / 128;
  atomicAdd(&aggF[(size_t)dst[e] * 128 + c], msg0[i]);
}

__global__ void k_cnt(const int* __restrict__ dst, float* __restrict__ cnt) {
  int e = blockIdx.x * blockDim.x + threadIdx.x;
  if (e < Ee) atomicAdd(&cnt[dst[e]], 1.f);
}

__global__ void k_h0x(const float* __restrict__ aggF, const float* __restrict__ cnt,
                      const float* __restrict__ FS, const float* __restrict__ h0,
                      float* __restrict__ x) {
  int i = blockIdx.x * blockDim.x + threadIdx.x;
  if (i >= Nn * 128) return;
  int o = i % 128, n = i / 128;
  float a = aggF[i] / fmaxf(cnt[n], 1.f);
  for (int m = 0; m < 32; ++m) a += FS[o * 32 + m] * h0[(size_t)n * 32 + m];
  x[i] = a;
}

__global__ void k_pool(const float* __restrict__ x, const int* __restrict__ gid,
                       float* __restrict__ gs) {
  int i = blockIdx.x * blockDim.x + threadIdx.x;
  if (i >= Nn * 128) return;
  int o = i % 128, n = i / 128;
  atomicAdd(&gs[(size_t)gid[n] * 128 + o], x[i]);
}

__global__ void k_gcnt(const int* __restrict__ gid, float* __restrict__ gc) {
  int n = blockIdx.x * blockDim.x + threadIdx.x;
  if (n < Nn) atomicAdd(&gc[gid[n]], 1.f);
}

__global__ __launch_bounds__(128)
void k_fc(const float* __restrict__ gs, const float* __restrict__ gc,
          const float* __restrict__ W1, const float* __restrict__ B1,
          const float* __restrict__ W2, const float* __restrict__ B2,
          float* __restrict__ out) {
  int g = blockIdx.x, j = threadIdx.x;
  __shared__ float pooled[128], red[128];
  pooled[j] = gs[(size_t)g * 128 + j] / fmaxf(gc[g], 1.f);
  __syncthreads();
  float a = B1[j];
  for (int i = 0; i < 128; ++i) a += pooled[i] * W1[i * 128 + j];
  red[j] = fmaxf(a, 0.f) * W2[j];
  __syncthreads();
  for (int s = 64; s > 0; s >>= 1) {
    if (j < s) red[j] += red[j + s];
    __syncthreads();
  }
  if (j == 0) out[g] = red[0] + B2[0];
}

// ------------------------------ host orchestration -------------------------

static inline int cdiv(int a, int b) { return (a + b - 1) / b; }

static void conv8(int di, int dO, const float* h2, const float* W3, const float* basis,
                  const float* hsrc, const int* esrc, float* msg, hipStream_t s) {
  dim3 g(Ee / 16), b(32);
#define CV8(DI, DO, NF) case (DI)*4 + (DO): \
    k_conv_pair<DI, DO, NF, 8><<<g, b, 0, s>>>(h2, W3, basis, hsrc, esrc, msg); break;
  switch (di * 4 + dO) {
    CV8(0, 0, 1) CV8(0, 1, 1) CV8(0, 2, 1) CV8(0, 3, 1)
    CV8(1, 0, 1) CV8(1, 1, 3) CV8(1, 2, 3) CV8(1, 3, 3)
    CV8(2, 0, 1) CV8(2, 1, 3) CV8(2, 2, 5) CV8(2, 3, 5)
    CV8(3, 0, 1) CV8(3, 1, 3) CV8(3, 2, 5) CV8(3, 3, 7)
  }
#undef CV8
}

static void conv128(int di, const float* h2, const float* W3, const float* basis,
                    const float* hsrc, const int* esrc, float* msg, hipStream_t s) {
  dim3 g(Ee / 16), b(32);
  switch (di) {
    case 0: k_conv_pair<0, 0, 1, 128><<<g, b, 0, s>>>(h2, W3, basis, hsrc, esrc, msg); break;
    case 1: k_conv_pair<1, 0, 1, 128><<<g, b, 0, s>>>(h2, W3, basis, hsrc, esrc, msg); break;
    case 2: k_conv_pair<2, 0, 1, 128><<<g, b, 0, s>>>(h2, W3, basis, hsrc, esrc, msg); break;
    case 3: k_conv_pair<3, 0, 1, 128><<<g, b, 0, s>>>(h2, W3, basis, hsrc, esrc, msg); break;
  }
}

extern "C" void kernel_launch(void* const* d_in, const int* in_sizes, int n_in,
                              void* d_out, int out_size, void* d_ws, size_t ws_size,
                              hipStream_t stream) {
  (void)in_sizes; (void)n_in; (void)out_size; (void)ws_size;

  const float* feat = (const float*)d_in[0];
  const float* ef   = (const float*)d_in[1];
  const float* rr   = (const float*)d_in[2];
  const int*   esrc = (const int*)d_in[3];
  const int*   edst = (const int*)d_in[4];
  const int*   gid  = (const int*)d_in[5];
  auto F      = [&](int i) { return (const float*)d_in[i]; };
  auto basisP = [&](int di, int dO) { return F(6 + di * 4 + dO); };

  // Param indexing assumes insertion-order flattening of the nested dicts.
  const int Lbase[4] = {22, 68, 252, 436};
  auto vpair = [&](int l, int dO, int di, int w) {
    return (l == 0) ? 22 + dO * 5 + w : Lbase[l] + (dO * 4 + di) * 5 + w; };
  auto kpair = [&](int l, int dO, int di, int w) {
    return (l == 0) ? 42 + w : Lbase[l] + 80 + (dO * 4 + di) * 5 + w; };
  auto qidx = [&](int l, int d) { return (l == 0) ? 47 : Lbase[l] + 160 + d; };
  auto pidx = [&](int l, int d) { return (l == 0) ? 48 + d : Lbase[l] + 164 + d; };
  auto nidx = [&](int l, int d, int w) {
    return ((l == 0) ? 52 : Lbase[l] + 168) + d * 4 + w; };
  const int FB = 620;  // final conv pairs; 640 final_self; 641/642 fc1; 643/644 fc2

  // ---- workspace carve (floats, 256B aligned) ----
  char* wp = (char*)d_ws;
  auto alloc = [&](size_t nf) {
    float* p = (float*)wp;
    wp += ((nf * 4 + 255) / 256) * 256;
    return p;
  };
  float* xe     = alloc((size_t)Ee * 5);
  float* h2     = alloc((size_t)Ee * 32);
  float* hA     = alloc((size_t)Nn * 32 * 16);
  float* hB     = alloc((size_t)Nn * 32 * 16);
  float* msgV   = alloc((size_t)Ee * 8 * 16);   // also final msg0 (E*128)
  float* msgK   = alloc((size_t)Ee * 8 * 16);
  float* qb     = alloc((size_t)Nn * 8 * 16);
  float* logits = alloc((size_t)Ee * 2);        // -> ex -> alpha (in place)
  float* mx     = alloc((size_t)Nn * 2);
  float* den    = alloc((size_t)Nn * 2);
  float* agg    = alloc((size_t)Nn * 8 * 16);
  float* aggF   = alloc((size_t)Nn * 128);
  float* cnt    = alloc((size_t)Nn);
  float* xbuf   = alloc((size_t)Nn * 128);
  float* gs     = alloc((size_t)Gg * 128);
  float* gc     = alloc((size_t)Gg);

  const int off[4] = {0, 1, 4, 9};  // cumsum of (2d+1)
  auto hAd  = [&](int d) { return hA + (size_t)Nn * 32 * off[d]; };
  auto hBd  = [&](int d) { return hB + (size_t)Nn * 32 * off[d]; };
  auto mVd  = [&](int d) { return msgV + (size_t)Ee * 8 * off[d]; };
  auto mKd  = [&](int d) { return msgK + (size_t)Ee * 8 * off[d]; };
  auto qd   = [&](int d) { return qb + (size_t)Nn * 8 * off[d]; };
  auto aggd = [&](int d) { return agg + (size_t)Nn * 8 * off[d]; };

  auto fill = [&](float* p, float v, int n) {
    k_fill<<<cdiv(n, 256), 256, 0, stream>>>(p, v, n);
  };
  auto radial = [&](int base) {
    k_radial_hidden<<<Ee / 128, 128, 0, stream>>>(xe, F(base), F(base + 1),
                                                  F(base + 2), F(base + 3), h2);
  };

  k_build_xe<<<cdiv(Ee, 256), 256, 0, stream>>>(rr, ef, xe);
  k_copy<<<cdiv(Nn * 32, 256), 256, 0, stream>>>(feat, hA, Nn * 32);

  for (int l = 0; l < 4; ++l) {
    const int nin = (l == 0) ? 1 : 4;

    // --- value conv (all 4 output degrees)
    fill(msgV, 0.f, Ee * 8 * 16);
    for (int dO = 0; dO < 4; ++dO)
      for (int di = 0; di < nin; ++di) {
        radial(vpair(l, dO, di, 0));
        conv8(di, dO, h2, F(vpair(l, dO, di, 4)), basisP(di, dO), hAd(di), esrc,
              mVd(dO), stream);
      }
    // --- key conv (output degrees = input degrees)
    fill(msgK, 0.f, Ee * 8 * 16);
    for (int dO = 0; dO < nin; ++dO)
      for (int di = 0; di < nin; ++di) {
        radial(kpair(l, dO, di, 0));
        conv8(di, dO, h2, F(kpair(l, dO, di, 4)), basisP(di, dO), hAd(di), esrc,
              mKd(dO), stream);
      }
    // --- queries, logits
    for (int d = 0; d < nin; ++d) {
      int M = 2 * d + 1, tot = Nn * 8 * M;
      k_q<<<cdiv(tot, 256), 256, 0, stream>>>(F(qidx(l, d)), hAd(d), qd(d), M, tot);
    }
    fill(logits, 0.f, Ee * 2);
    for (int d = 0; d < nin; ++d)
      k_logits<<<cdiv(Ee * 2, 256), 256, 0, stream>>>(mKd(d), qd(d), edst, logits,
                                                      2 * d + 1);
    const float invs = 1.f / sqrtf((l == 0) ? 8.f : 128.f);
    k_scale<<<cdiv(Ee * 2, 256), 256, 0, stream>>>(logits, invs, Ee * 2);

    // --- segment softmax over dst
    fill(mx, -1e30f, Nn * 2);
    fill(den, 0.f, Nn * 2);
    k_segmax<<<cdiv(Ee * 2, 256), 256, 0, stream>>>(logits, edst, mx);
    k_expsum<<<cdiv(Ee * 2, 256), 256, 0, stream>>>(logits, edst, mx, den);
    k_alpha<<<cdiv(Ee * 2, 256), 256, 0, stream>>>(logits, edst, den);

    // --- weighted aggregation + projection + residual + gated norm
    fill(agg, 0.f, Nn * 8 * 16);
    for (int d = 0; d < 4; ++d) {
      int M = 2 * d + 1, tot = Ee * 8 * M;
      k_aggv<<<cdiv(tot, 256), 256, 0, stream>>>(mVd(d), logits, edst, aggd(d), M, tot);
    }
    for (int d = 0; d < 4; ++d) {
      int M = 2 * d + 1, tot = Nn * 32 * M;
      k_proj<<<cdiv(tot, 256), 256, 0, stream>>>(F(pidx(l, d)), aggd(d),
                                                 (d < nin) ? hAd(d) : nullptr,
                                                 hBd(d), M, tot);
    }
    for (int d = 0; d < 4; ++d)
      k_gnorm<<<Nn, 32, 0, stream>>>(F(nidx(l, d, 0)), F(nidx(l, d, 1)),
                                     F(nidx(l, d, 2)), F(nidx(l, d, 3)),
                                     hBd(d), hAd(d), 2 * d + 1);
  }

  // --- final conv (degree 0, 128 channels), reuse msgV as (E,128)
  float* msg0 = msgV;
  fill(msg0, 0.f, Ee * 128);
  for (int di = 0; di < 4; ++di) {
    radial(FB + di * 5);
    conv128(di, h2, F(FB + di * 5 + 4), basisP(di, 0), hAd(di), esrc, msg0, stream);
  }
  fill(aggF, 0.f, Nn * 128);
  fill(cnt, 0.f, Nn);
  k_faggr<<<cdiv(Ee * 128, 256), 256, 0, stream>>>(msg0, edst, aggF);
  k_cnt<<<cdiv(Ee, 256), 256, 0, stream>>>(edst, cnt);
  k_h0x<<<cdiv(Nn * 128, 256), 256, 0, stream>>>(aggF, cnt, F(640), hAd(0), xbuf);

  fill(gs, 0.f, Gg * 128);
  fill(gc, 0.f, Gg);
  k_pool<<<cdiv(Nn * 128, 256), 256, 0, stream>>>(xbuf, gid, gs);
  k_gcnt<<<cdiv(Nn, 256), 256, 0, stream>>>(gid, gc);
  k_fc<<<Gg, 128, 0, stream>>>(gs, gc, F(641), F(642), F(643), F(644), (float*)d_out);
}
